// NDDE_1D_81690277970385
// MI455X (gfx1250) — matrix operand order repaired and checked by reading, not compiled, not run on hardware
//
#include <hip/hip_runtime.h>

#define Dd   256
#define Hh   1024
#define NTAU 10
#define NWG  16          // persistent workgroups; 64 W1-rows / 64 W2-cols each
#define TPB  256         // 8 waves (wave32)
#define RW1  (Hh / NWG)  // 64
#define KW1  (2 * Dd)    // 512

typedef __bf16 bf16;
typedef __attribute__((ext_vector_type(16))) __bf16 v16bf;
typedef __attribute__((ext_vector_type(8)))  __bf16 v8bf;
typedef __attribute__((ext_vector_type(8)))  float  v8f;

union AFrag { v16bf v; v8bf h[2]; };

// A-matrix (16x32 bf16) fragment: this lane's 16 elements are two contiguous
// 8-element (16B) pieces at K-offsets +0 and +8 from `p` (caller already
// applied the +16 shift for lanes 16..31).  Contiguous k = 16h+s order, not ISA table; B matches.
__device__ __forceinline__ v16bf load_a(const bf16* p) {
  AFrag a;
  a.h[0] = *(const v8bf*)(p);
  a.h[1] = *(const v8bf*)(p + 8);
  return a.v;
}

__device__ __forceinline__ v8f wmma3(v8f acc, v16bf ah, v16bf al, v16bf bh, v16bf bl) {
  // (Whi+Wlo)(zhi+zlo) ~= Whi*zhi + Whi*zlo + Wlo*zhi  (f32 accumulate)
  acc = __builtin_amdgcn_wmma_f32_16x16x32_bf16(false, ah, false, bh, (short)0, acc, false, false);
  acc = __builtin_amdgcn_wmma_f32_16x16x32_bf16(false, ah, false, bl, (short)0, acc, false, false);
  acc = __builtin_amdgcn_wmma_f32_16x16x32_bf16(false, al, false, bh, (short)0, acc, false, false);
  return acc;
}

// One-time split of fp32 weights into bf16 hi + bf16 lo residual.
__global__ void ndde_prep_split(const float* __restrict__ W1, const float* __restrict__ W2,
                                bf16* __restrict__ w1hi, bf16* __restrict__ w1lo,
                                bf16* __restrict__ w2hi, bf16* __restrict__ w2lo) {
  int i = blockIdx.x * blockDim.x + threadIdx.x;
  const int n1 = Hh * KW1;            // 524288
  const int nt = n1 + Dd * Hh;        // + 262144 = 786432 (exact grid)
  if (i >= nt) return;
  float w; bf16* hi; bf16* lo; int j;
  if (i < n1) { j = i;      w = W1[j]; hi = w1hi; lo = w1lo; }
  else        { j = i - n1; w = W2[j]; hi = w2hi; lo = w2lo; }
  bf16 h = (bf16)w;
  hi[j] = h;
  lo[j] = (bf16)(w - (float)h);
}

__global__ void ndde_init_bar(unsigned* bar) { *bar = 0u; }

__global__ __launch_bounds__(TPB) void ndde_persistent(
    const float* __restrict__ x0,  const float* __restrict__ b1,
    const float* __restrict__ b2,  const int*   __restrict__ tauPtr,
    const bf16* __restrict__ w1hi, const bf16* __restrict__ w1lo,
    const bf16* __restrict__ w2hi, const bf16* __restrict__ w2lo,
    float* __restrict__ part, unsigned* __restrict__ bar,
    float* __restrict__ out, int N, int Ncol) {

  __shared__ float xls[Dd];                 // current state (replicated per WG)
  __shared__ float ring[NTAU][Dd];          // delay buffer  (replicated per WG)
  __shared__ __align__(32) bf16 zhi[KW1], zlo[KW1];   // concat(x,y) hi/lo
  __shared__ __align__(32) bf16 hhi[RW1], hlo[RW1];   // tanh hidden slice hi/lo
  __shared__ float hpart[2][RW1];           // K-half partials for W1 matvec

  const int tid  = threadIdx.x;             // == state index d (Dd == TPB)
  const int wg   = blockIdx.x;
  const int wave = tid >> 5;
  const int lane = tid & 31;
  const bool hiHalf = lane >= 16;
  const int  mrow   = lane & 15;

  const float dt  = 0.1f * (float)tauPtr[0];
  const float rb2 = b2[tid];
  const float rb1 = (tid < RW1) ? b1[wg * RW1 + tid] : 0.0f;

  xls[tid] = x0[tid];
  #pragma unroll
  for (int j = 0; j < NTAU; ++j) ring[j][tid] = x0[tid];
  if (wg == 0) out[(size_t)tid * Ncol] = x0[tid];   // column 0 = x_0
  __syncthreads();

  // Phase-1 assignment: wave -> (row group of 16, K half of 256)
  const int rg = wave & 3;
  const int kh = wave >> 2;
  const int row1 = wg * RW1 + rg * 16 + mrow;
  const bf16* r1h = w1hi + (size_t)row1 * KW1;
  const bf16* r1l = w1lo + (size_t)row1 * KW1;

  int head = 0;
  for (int t = 0; t < N; ++t) {
    // ---- build z = concat(x, x(t - tau)) as bf16 hi/lo ----
    float xv = xls[tid];
    float yv = ring[head][tid];
    bf16 xh = (bf16)xv;  zhi[tid]      = xh;  zlo[tid]      = (bf16)(xv - (float)xh);
    bf16 yh = (bf16)yv;  zhi[Dd + tid] = yh;  zlo[Dd + tid] = (bf16)(yv - (float)yh);
    __syncthreads();

    // ---- Phase 1: h_part = W1_slice @ z (WMMA bf16x2, f32 accum) ----
    {
      v8f acc = {};
      #pragma unroll
      for (int c = 0; c < 8; ++c) {
        int cc = kh * 8 + c;                       // K chunk of 32
        int ae = cc * 32 + (hiHalf ? 16 : 0);
        v16bf ah = load_a(r1h + ae);
        v16bf al = load_a(r1l + ae);
        int be = cc * 32 + (hiHalf ? 16 : 0);
        v16bf bh = *(const v16bf*)&zhi[be];
        v16bf bl = *(const v16bf*)&zlo[be];
        acc = wmma3(acc, ah, al, bh, bl);
      }
      // C layout: VGPR j -> (M=j, N=lane) lanes 0-15, (M=8+j) lanes 16-31.
      // All columns equal (B column-broadcast) -> read column 0 via lanes 0/16.
      if (lane == 0) {
        #pragma unroll
        for (int j = 0; j < 8; ++j) hpart[kh][rg * 16 + j] = acc[j];
      } else if (lane == 16) {
        #pragma unroll
        for (int j = 0; j < 8; ++j) hpart[kh][rg * 16 + 8 + j] = acc[j];
      }
    }
    __syncthreads();

    if (tid < RW1) {
      float hv = tanhf(hpart[0][tid] + hpart[1][tid] + rb1);
      bf16 hh = (bf16)hv;
      hhi[tid] = hh;  hlo[tid] = (bf16)(hv - (float)hh);
    }
    __syncthreads();

    // ---- Phase 2: partial out = W2[:, wg-cols] @ h_slice ----
    {
      v8f a0 = {}, a1 = {};
      const int r0 = wave * 32 + mrow;
      #pragma unroll
      for (int cg = 0; cg < 2; ++cg) {
        int be = cg * 32 + (hiHalf ? 16 : 0);
        v16bf bh = *(const v16bf*)&hhi[be];
        v16bf bl = *(const v16bf*)&hlo[be];
        int col = wg * RW1 + cg * 32 + (hiHalf ? 16 : 0);
        a0 = wmma3(a0, load_a(w2hi + (size_t)r0 * Hh + col),
                       load_a(w2lo + (size_t)r0 * Hh + col), bh, bl);
        a1 = wmma3(a1, load_a(w2hi + (size_t)(r0 + 16) * Hh + col),
                       load_a(w2lo + (size_t)(r0 + 16) * Hh + col), bh, bl);
      }
      float* pp = part + ((size_t)(t & 1) * NWG + wg) * Dd;   // double-buffered
      if (lane == 0) {
        #pragma unroll
        for (int j = 0; j < 8; ++j) {
          pp[wave * 32 + j]      = a0[j];
          pp[wave * 32 + 16 + j] = a1[j];
        }
      } else if (lane == 16) {
        #pragma unroll
        for (int j = 0; j < 8; ++j) {
          pp[wave * 32 + 8 + j]  = a0[j];
          pp[wave * 32 + 24 + j] = a1[j];
        }
      }
    }
    __threadfence();     // release partials to device scope
    __syncthreads();

    // ---- one device-wide barrier per step (monotonic counter) ----
    if (tid == 0) {
      __hip_atomic_fetch_add(bar, 1u, __ATOMIC_RELEASE, __HIP_MEMORY_SCOPE_AGENT);
      const unsigned target = (unsigned)NWG * (unsigned)(t + 1);
      while (__hip_atomic_load(bar, __ATOMIC_ACQUIRE, __HIP_MEMORY_SCOPE_AGENT) < target)
        __builtin_amdgcn_s_sleep(1);
    }
    __syncthreads();
    __threadfence();     // acquire: invalidate WGP$ so partial reads are fresh

    // ---- deterministic fixed-order reduction + Euler update (replicated) ----
    float s = 0.0f;
    const float* pb = part + (size_t)(t & 1) * NWG * Dd + tid;
    #pragma unroll
    for (int g = 0; g < NWG; ++g) s += pb[g * Dd];
    float xn = xv + dt * (s + rb2);
    ring[head][tid] = xv;          // push pre-update state into delay buffer
    xls[tid] = xn;
    if (wg == 0) out[(size_t)tid * Ncol + (t + 1)] = xn;
    head = head + 1; if (head == NTAU) head = 0;
  }
}

extern "C" void kernel_launch(void* const* d_in, const int* in_sizes, int n_in,
                              void* d_out, int out_size, void* d_ws, size_t ws_size,
                              hipStream_t stream) {
  (void)in_sizes; (void)n_in; (void)ws_size;
  const float* x0 = (const float*)d_in[0];
  const float* W1 = (const float*)d_in[1];
  const float* b1 = (const float*)d_in[2];
  const float* W2 = (const float*)d_in[3];
  const float* b2 = (const float*)d_in[4];
  const int*  tau = (const int*)d_in[5];
  // d_in[6] = T (unused: N derived from out_size on host)

  const int Ncol = out_size / Dd;   // N + 1 columns
  const int N    = Ncol - 1;

  char* ws = (char*)d_ws;
  bf16*     w1hi = (bf16*)(ws + 0);
  bf16*     w1lo = (bf16*)(ws + (1u << 20));
  bf16*     w2hi = (bf16*)(ws + (2u << 20));
  bf16*     w2lo = (bf16*)(ws + (2u << 20) + (512u << 10));
  float*    part = (float*)(ws + (3u << 20));                 // 2*NWG*Dd f32
  unsigned* bar  = (unsigned*)(ws + (3u << 20) + (64u << 10));

  ndde_prep_split<<<(Hh * KW1 + Dd * Hh + TPB - 1) / TPB, TPB, 0, stream>>>(
      W1, W2, w1hi, w1lo, w2hi, w2lo);
  ndde_init_bar<<<1, 1, 0, stream>>>(bar);
  ndde_persistent<<<NWG, TPB, 0, stream>>>(
      x0, b1, b2, tau, w1hi, w1lo, w2hi, w2lo, part, bar,
      (float*)d_out, N, Ncol);
}
